// TPro_42872363548904
// MI455X (gfx1250) — compile-verified, matrix operand order checked by reading, not run on hardware
//
#include <hip/hip_runtime.h>
#include <hip/hip_bf16.h>

typedef __attribute__((ext_vector_type(16))) __bf16 v16bf;
typedef __attribute__((ext_vector_type(8)))  float  v8f;

#define BDIM 8
#define DDIM 64
#define NHEAD 8
#define HD 8
#define SEQ 128
#define OUTL 128
#define HW 1024                      // 32*32
#define NPOS (BDIM*OUTL*HW)          // 1,048,576 positions (b,o,h,w)

// ---------------------------------------------------------------------------
// Kernel 0: zero the 512-float stats region at the front of the workspace.
// ---------------------------------------------------------------------------
__global__ void TPro_zero_stats(float* __restrict__ stats) {
    stats[blockIdx.x * 256 + threadIdx.x] = 0.0f;
}

// ---------------------------------------------------------------------------
// Kernel 1: per-head seq-Linear GEMM (bf16 WMMA, f32 acc) + BN1 sum/sumsq.
//   A = x[(bhw)*64 + n*8+hd][s]  (row-major, s contiguous)
//   B[s][o] = W_heads[n][o][s]
//   Writes Q[pos*64 + d], pos = ((b*128)+o)*1024 + hw, d = n*8+hd
// One wave = one 16-row M-tile x all 8 O-tiles (A fragment reused 8x).
// ---------------------------------------------------------------------------
__global__ __launch_bounds__(256) void TPro_gemm1_bn1(
    const float* __restrict__ x, const float* __restrict__ Wh,
    const float* __restrict__ bh, float* __restrict__ Q,
    float* __restrict__ sum1, float* __restrict__ sq1)
{
    __shared__ float s_sum[DDIM], s_sq[DDIM];
    const int tid = threadIdx.x;
    if (tid < DDIM) { s_sum[tid] = 0.0f; s_sq[tid] = 0.0f; }
    __syncthreads();

    const int lane = tid & 31;
    const int gw   = blockIdx.x * 8 + (tid >> 5);  // 0..32767
    const int n      = gw >> 12;                   // head 0..7
    const int m_tile = gw & 4095;                  // 0..4095 (65536 rows/head)
    const int l16 = lane & 15;
    const int h16 = lane >> 4;

    // A row owned by this lane (per-head row m = bhw*8 + hd)
    const int m  = m_tile * 16 + l16;
    const int p  = m >> 3;                         // bhw index
    const int hd = m & 7;
    const float* __restrict__ arow = x + ((size_t)p * DDIM + n * HD + hd) * SEQ;

    v8f acc[8];
    #pragma unroll
    for (int ot = 0; ot < 8; ++ot)
        #pragma unroll
        for (int e = 0; e < 8; ++e) acc[ot][e] = 0.0f;

    for (int k0 = 0; k0 < SEQ; k0 += 32) {
        // A fragment: lane holds K = {h16*8..+7} and {16+h16*8..+7} of row M=l16
        v16bf a;
        const float* a0 = arow + k0 + h16 * 8;
        #pragma unroll
        for (int e = 0; e < 8; ++e) { a[e] = (__bf16)a0[e]; a[8 + e] = (__bf16)a0[16 + e]; }
        #pragma unroll
        for (int ot = 0; ot < 8; ++ot) {
            const int o = ot * 16 + l16;
            const float* b0 = Wh + ((size_t)n * OUTL + o) * SEQ + k0 + h16 * 8;
            v16bf b;
            #pragma unroll
            for (int e = 0; e < 8; ++e) { b[e] = (__bf16)b0[e]; b[8 + e] = (__bf16)b0[16 + e]; }
            acc[ot] = __builtin_amdgcn_wmma_f32_16x16x32_bf16(
                false, a, false, b, (short)0, acc[ot], false, false);
        }
    }

    // Epilogue: lane's 8 C slots = channels d=n*8+j at ONE spatial pos pr.
    const float inv = 0.088388347648318447f;       // 1/sqrt(128)
    const int pr  = m_tile * 2 + h16;              // bhw for rows 8*h16..+7
    const int b_  = pr >> 10;
    const int hw  = pr & 1023;

    float psum[8], psq[8];
    #pragma unroll
    for (int j = 0; j < 8; ++j) { psum[j] = 0.0f; psq[j] = 0.0f; }

    #pragma unroll
    for (int ot = 0; ot < 8; ++ot) {
        const int o = ot * 16 + l16;
        const float bias = bh[n * OUTL + o];
        float tmp[8];
        #pragma unroll
        for (int j = 0; j < 8; ++j) {
            float v = acc[ot][j] * inv + bias;
            tmp[j] = v; psum[j] += v; psq[j] += v * v;
        }
        float* qp = Q + (((size_t)b_ * OUTL + o) * HW + hw) * DDIM + n * HD;
        *(float4*)(qp)     = make_float4(tmp[0], tmp[1], tmp[2], tmp[3]);
        *(float4*)(qp + 4) = make_float4(tmp[4], tmp[5], tmp[6], tmp[7]);
    }

    // BN1 stats: wave reduce, one LDS atomic per channel, block flush.
    #pragma unroll
    for (int j = 0; j < 8; ++j) {
        float s = psum[j], q = psq[j];
        #pragma unroll
        for (int off = 16; off > 0; off >>= 1) {
            s += __shfl_xor(s, off, 32);
            q += __shfl_xor(q, off, 32);
        }
        if (lane == 0) {
            atomicAdd(&s_sum[n * HD + j], s);
            atomicAdd(&s_sq[n * HD + j], q);
        }
    }
    __syncthreads();
    if (tid < DDIM) { atomicAdd(&sum1[tid], s_sum[tid]); atomicAdd(&sq1[tid], s_sq[tid]); }
}

// ---------------------------------------------------------------------------
// Kernel 2/4: fold BN stats into scale/shift.
// ---------------------------------------------------------------------------
__global__ void TPro_finalize_bn(
    const float* __restrict__ sum, const float* __restrict__ sq,
    const float* __restrict__ gamma, const float* __restrict__ beta,
    float* __restrict__ scale, float* __restrict__ shift, float invN)
{
    const int d = threadIdx.x;                     // 64 threads
    const float mean = sum[d] * invN;
    const float var  = sq[d] * invN - mean * mean;
    const float sc   = gamma[d] * rsqrtf(var + 1e-5f);
    scale[d] = sc;
    shift[d] = beta[d] - mean * sc;
}

// ---------------------------------------------------------------------------
// Kernel 3: channel GEMM (conv 1x1x1) with BN1+ReLU fused into the A reads,
// bf16 WMMA, raw y -> d_out in [b,k,o,h,w] layout, BN2 sum/sumsq accumulated.
// One wave = one 16-row pos-tile x all 4 K-out tiles.
// ---------------------------------------------------------------------------
__global__ __launch_bounds__(256) void TPro_gemm2_bn2(
    const float* __restrict__ Q, const float* __restrict__ cw,
    const float* __restrict__ cb, const float* __restrict__ scale1,
    const float* __restrict__ shift1, float* __restrict__ out,
    float* __restrict__ sum2, float* __restrict__ sq2)
{
    __shared__ float s_scale[DDIM], s_shift[DDIM], s_sum[DDIM], s_sq[DDIM];
    const int tid = threadIdx.x;
    if (tid < DDIM) {
        s_scale[tid] = scale1[tid]; s_shift[tid] = shift1[tid];
        s_sum[tid] = 0.0f; s_sq[tid] = 0.0f;
    }
    __syncthreads();

    const int lane = tid & 31;
    const int gw   = blockIdx.x * 8 + (tid >> 5);  // m2 tile, 0..65535
    const int l16  = lane & 15;
    const int h16  = lane >> 4;

    const int pos = gw * 16 + l16;
    const float* __restrict__ arow = Q + (size_t)pos * DDIM;

    v8f acc[4];
    #pragma unroll
    for (int nt = 0; nt < 4; ++nt)
        #pragma unroll
        for (int e = 0; e < 8; ++e) acc[nt][e] = 0.0f;

    #pragma unroll
    for (int k0 = 0; k0 < DDIM; k0 += 32) {
        const int cbase = k0 + h16 * 8;
        v16bf a;
        #pragma unroll
        for (int e = 0; e < 8; ++e) {
            const int c0 = cbase + e, c1 = cbase + 16 + e;
            const float v0 = fmaxf(0.0f, arow[c0] * s_scale[c0] + s_shift[c0]);
            const float v1 = fmaxf(0.0f, arow[c1] * s_scale[c1] + s_shift[c1]);
            a[e] = (__bf16)v0; a[8 + e] = (__bf16)v1;
        }
        #pragma unroll
        for (int nt = 0; nt < 4; ++nt) {
            const int k = nt * 16 + l16;
            const float* b0 = cw + (size_t)k * DDIM + cbase;  // B[c][k] = conv_w[k][c]
            v16bf b;
            #pragma unroll
            for (int e = 0; e < 8; ++e) { b[e] = (__bf16)b0[e]; b[8 + e] = (__bf16)b0[16 + e]; }
            acc[nt] = __builtin_amdgcn_wmma_f32_16x16x32_bf16(
                false, a, false, b, (short)0, acc[nt], false, false);
        }
    }

    // Rows j=0..7 are 8 consecutive positions -> 8 consecutive hw in output.
    const int pbase = gw * 16 + h16 * 8;
    const int b_  = pbase >> 17;
    const int o   = (pbase >> 10) & 127;
    const int hwb = pbase & 1023;

    #pragma unroll
    for (int nt = 0; nt < 4; ++nt) {
        const int k = nt * 16 + l16;
        const float bias = cb[k];
        float tmp[8], ls = 0.0f, lq = 0.0f;
        #pragma unroll
        for (int j = 0; j < 8; ++j) {
            const float v = acc[nt][j] + bias;
            tmp[j] = v; ls += v; lq += v * v;
        }
        float* op = out + (((size_t)b_ * DDIM + k) * OUTL + o) * HW + hwb;
        *(float4*)(op)     = make_float4(tmp[0], tmp[1], tmp[2], tmp[3]);
        *(float4*)(op + 4) = make_float4(tmp[4], tmp[5], tmp[6], tmp[7]);
        atomicAdd(&s_sum[k], ls);
        atomicAdd(&s_sq[k], lq);
    }
    __syncthreads();
    if (tid < DDIM) { atomicAdd(&sum2[tid], s_sum[tid]); atomicAdd(&sq2[tid], s_sq[tid]); }
}

// ---------------------------------------------------------------------------
// Kernel 5: in-place BN2 + ReLU sweep over the output (float4 vectorized).
// ---------------------------------------------------------------------------
__global__ __launch_bounds__(256) void TPro_bn2_apply(
    float* __restrict__ out, const float* __restrict__ scale,
    const float* __restrict__ shift)
{
    const size_t i = (size_t)blockIdx.x * blockDim.x + threadIdx.x;  // float4 idx
    float4 v = ((float4*)out)[i];
    const int k = (int)((i * 4) >> 17) & 63;   // 131072 contiguous elems per (b,k)
    const float sc = scale[k], sh = shift[k];
    v.x = fmaxf(0.0f, v.x * sc + sh);
    v.y = fmaxf(0.0f, v.y * sc + sh);
    v.z = fmaxf(0.0f, v.z * sc + sh);
    v.w = fmaxf(0.0f, v.w * sc + sh);
    ((float4*)out)[i] = v;
}

// ---------------------------------------------------------------------------
extern "C" void kernel_launch(void* const* d_in, const int* in_sizes, int n_in,
                              void* d_out, int out_size, void* d_ws, size_t ws_size,
                              hipStream_t stream)
{
    const float* x   = (const float*)d_in[0];
    const float* Wh  = (const float*)d_in[1];
    const float* bh  = (const float*)d_in[2];
    const float* g1  = (const float*)d_in[3];
    const float* be1 = (const float*)d_in[4];
    const float* cw  = (const float*)d_in[5];
    const float* cb  = (const float*)d_in[6];
    const float* g2  = (const float*)d_in[7];
    const float* be2 = (const float*)d_in[8];
    float* out = (float*)d_out;
    float* wsf = (float*)d_ws;

    float* sum1 = wsf + 0;   float* sq1 = wsf + 64;
    float* sc1  = wsf + 128; float* sh1 = wsf + 192;
    float* sum2 = wsf + 256; float* sq2 = wsf + 320;
    float* sc2  = wsf + 384; float* sh2 = wsf + 448;
    float* Q    = wsf + 1024;            // 256 MB intermediate [pos][d]

    const float invN = 1.0f / (float)NPOS;

    TPro_zero_stats <<<2, 256, 0, stream>>>(wsf);
    TPro_gemm1_bn1  <<<4096, 256, 0, stream>>>(x, Wh, bh, Q, sum1, sq1);
    TPro_finalize_bn<<<1, 64, 0, stream>>>(sum1, sq1, g1, be1, sc1, sh1, invN);
    TPro_gemm2_bn2  <<<8192, 256, 0, stream>>>(Q, cw, cb, sc1, sh1, out, sum2, sq2);
    TPro_finalize_bn<<<1, 64, 0, stream>>>(sum2, sq2, g2, be2, sc2, sh2, invN);
    TPro_bn2_apply  <<<65536, 256, 0, stream>>>(out, sc2, sh2);
}
